// UnidirectionalMultiheadAttention_46231027974723
// MI455X (gfx1250) — matrix operand order repaired and checked by reading, not compiled, not run on hardware
//
#include <hip/hip_runtime.h>

typedef _Float16 h16;
typedef __attribute__((ext_vector_type(16))) _Float16 v16h;
typedef __attribute__((ext_vector_type(8)))  _Float16 v8h;
typedef __attribute__((ext_vector_type(8)))  float    v8f;

#define B_  2
#define S_  2048
#define D_  1024
#define H_  16
#define HD_ 64
#define M_  (B_ * S_)   // 4096 tokens

// ---------------------------------------------------------------------------
// CDNA5 async copy to LDS (ASYNCcnt-tracked), per-lane 16B
// ---------------------------------------------------------------------------
__device__ __forceinline__ void async_b128(const h16* gsrc, h16* lds_dst) {
  unsigned l = (unsigned)(unsigned long long)(const void*)lds_dst;
  asm volatile("global_load_async_to_lds_b128 %0, %1, off"
               :: "v"(l), "v"(gsrc)
               : "memory");
}
template <int N>
__device__ __forceinline__ void wait_asynccnt() {
  asm volatile("s_wait_asynccnt %0" :: "i"(N) : "memory");
}

// ---------------------------------------------------------------------------
// WMMA helpers (CDNA5: D = A(16x32 f16) * B(32x16 f16) + C(16x16 f32))
// ---------------------------------------------------------------------------
__device__ __forceinline__ v8f wmma16(v16h a, v16h b, v8f c) {
  return __builtin_amdgcn_wmma_f32_16x16x32_f16(false, a, false, b, (short)0, c,
                                                false, false);
}

// A fragment: 16(M) x 32(K) f16 from row-major [rows][ld].
__device__ __forceinline__ v16h frag_a(const h16* base, int ld, int row0, int k0) {
  const int lane = threadIdx.x & 31;
  const int m  = lane & 15;
  const int kh = (lane >> 4) << 3;                 // 0 or 8
  const h16* p = base + (size_t)(row0 + m) * ld + k0 + kh;
  v8h lo = *(const v8h*)p;
  v8h hh = *(const v8h*)(p + 16);
  v16h r;
#pragma unroll
  for (int i = 0; i < 8; ++i) { r[i] = lo[i]; r[i + 8] = hh[i]; }
  return r;
}

// B fragment: 32(K) x 16(N) f16, source stored TRANSPOSED as T[N][K] row-major.
__device__ __forceinline__ v16h frag_b(const h16* T, int ld, int k0, int n0) {
  const int lane = threadIdx.x & 31;
  const int n  = lane & 15;
  const int ko = (lane >> 4) << 3;                 // 0 or 8
  const h16* p = T + (size_t)(n0 + n) * ld + k0 + ko;
  v8h lo = *(const v8h*)p;
  v8h hh = *(const v8h*)(p + 16);
  v16h r;
#pragma unroll
  for (int i = 0; i < 8; ++i) { r[i] = lo[i]; r[i + 8] = hh[i]; }
  return r;
}

// GEMM compute phase: 32 rows x 64 cols, one staged 64x32 B panel in LDS.
// 2 A-fragments (global), 4 B-fragments (LDS, register-reused) -> 8 WMMAs.
__device__ __forceinline__ void gemm_phase(const h16* A, int row0, int kk,
                                           const h16* Bl, v8f acc[8]) {
  v16h a0 = frag_a(A, D_, row0, kk);
  v16h a1 = frag_a(A, D_, row0 + 16, kk);
  v16h b0 = frag_b(Bl, 32, 0, 0);
  v16h b1 = frag_b(Bl, 32, 0, 16);
  v16h b2 = frag_b(Bl, 32, 0, 32);
  v16h b3 = frag_b(Bl, 32, 0, 48);
  acc[0] = wmma16(a0, b0, acc[0]);
  acc[1] = wmma16(a0, b1, acc[1]);
  acc[2] = wmma16(a0, b2, acc[2]);
  acc[3] = wmma16(a0, b3, acc[3]);
  acc[4] = wmma16(a1, b0, acc[4]);
  acc[5] = wmma16(a1, b1, acc[5]);
  acc[6] = wmma16(a1, b2, acc[6]);
  acc[7] = wmma16(a1, b3, acc[7]);
}

// ---------------------------------------------------------------------------
// Prep: f32 -> f16 convert, and [K][N] f32 -> [N][K] f16 transpose
// ---------------------------------------------------------------------------
__global__ __launch_bounds__(256) void cvt_f16(const float* __restrict__ src,
                                               h16* __restrict__ dst, int nElem) {
  int i = blockIdx.x * blockDim.x + threadIdx.x;
  if (i < nElem) dst[i] = (h16)src[i];
}

__global__ __launch_bounds__(256) void transpose_cvt(const float* __restrict__ W,
                                                     h16* __restrict__ Wt,
                                                     int K, int N) {
  int o = blockIdx.x * blockDim.x + threadIdx.x;
  if (o < K * N) {
    int nn = o / K, kk = o % K;
    Wt[o] = (h16)W[(size_t)kk * N + nn];
  }
}

// ---------------------------------------------------------------------------
// QKV projection: [4096 x 1024] x [1024 x 3072] + bias, scattered into
// Q[b,h,s,e], K[b,h,s,e], V^T[b,h,e,s] (all f16).
// Block = 256 rows x 64 cols; each of 8 waves owns 32 rows. The 64x32 weight
// panel is double-buffered in LDS with async-to-LDS copies; the k-loop is
// manually unrolled x2 so buffer indices are compile-time constants.
// ---------------------------------------------------------------------------
__global__ __launch_bounds__(256) void qkv_gemm(const h16* __restrict__ xh,
                                                const h16* __restrict__ Wt,
                                                const float* __restrict__ bias,
                                                h16* __restrict__ Qh,
                                                h16* __restrict__ Kh,
                                                h16* __restrict__ Vt) {
  __shared__ __align__(16) h16 BtL[2][64 * 32];
  const int t    = threadIdx.x;
  const int lane = t & 31;
  const int w    = t >> 5;
  const int NB   = (3 * D_) / 64;                  // 48 column panels
  const int blkM = blockIdx.x / NB, blkN = blockIdx.x % NB;
  const int row0 = blkM * 256 + w * 32, col0 = blkN * 64;
  const int n = lane & 15, hi = lane >> 4;
  const int rr = t >> 2, piece = (t & 3) * 8;      // staging: 64 rows x 4x16B
  const h16* Wrow = Wt + (size_t)(col0 + rr) * D_ + piece;

  async_b128(Wrow, &BtL[0][rr * 32 + piece]);

  v8f acc[8] = {};
  for (int kk = 0; kk < D_; kk += 64) {
    // phase 0: compute panel kk from buf0, prefetch kk+32 -> buf1 (always valid)
    async_b128(Wrow + (kk + 32), &BtL[1][rr * 32 + piece]);
    wait_asynccnt<1>();
    __syncthreads();
    __builtin_prefetch(xh + (size_t)(row0 + (lane & 15)) * D_ + kk + 64, 0, 3);
    gemm_phase(xh, row0, kk, &BtL[0][0], acc);
    __syncthreads();
    // phase 1: compute panel kk+32 from buf1, prefetch kk+64 -> buf0
    if (kk + 64 < D_) {
      async_b128(Wrow + (kk + 64), &BtL[0][rr * 32 + piece]);
      wait_asynccnt<1>();
    } else {
      wait_asynccnt<0>();
    }
    __syncthreads();
    gemm_phase(xh, row0, kk + 32, &BtL[1][0], acc);
    __syncthreads();
  }

#pragma unroll
  for (int g = 0; g < 2; ++g) {
    const int rbase = row0 + g * 16;
    const int b     = rbase >> 11;                 // token / S
    const int sl0   = (rbase & (S_ - 1)) + hi * 8;
#pragma unroll
    for (int f = 0; f < 4; ++f) {
      const v8f& a   = acc[g * 4 + f];
      const int col  = col0 + f * 16 + n;
      const float bv = bias[col];
      const int part = col >> 10;                  // 0=Q 1=K 2=V
      const int rem  = col & (D_ - 1);
      const int hh   = rem >> 6, e = rem & 63;
      const int bh   = b * H_ + hh;
      if (part == 2) {                             // V stored transposed [e][s]
        v8h tv;
#pragma unroll
        for (int r = 0; r < 8; ++r) tv[r] = (h16)(a[r] + bv);
        *(v8h*)(Vt + ((size_t)(bh * HD_ + e)) * S_ + sl0) = tv;
      } else {
        h16* dst = (part == 0 ? Qh : Kh) + ((size_t)bh * S_ + sl0) * HD_ + e;
#pragma unroll
        for (int r = 0; r < 8; ++r) dst[(size_t)r * HD_] = (h16)(a[r] + bv);
      }
    }
  }
}

// ---------------------------------------------------------------------------
// One causal flash-attention chunk (32 keys) for one wave's 16 queries.
// Scores are the D-matrix [M=key, N=query]; softmax stats are per-lane.
// ---------------------------------------------------------------------------
__device__ __forceinline__ void flash_chunk(
    const h16* Kc, const h16* Vc, int k0, v16h qb0, v16h qb1,
    int qidx, int hi, int n, h16* pw_base, float* bb,
    v8f& o0, v8f& o1, v8f& o2, v8f& o3, float& m_run, float& l_run) {
  const int lane = threadIdx.x & 31;
  v8f p0 = {}, p1 = {};
  p0 = wmma16(frag_a(Kc, HD_, 0,  0),  qb0, p0);
  p0 = wmma16(frag_a(Kc, HD_, 0,  32), qb1, p0);
  p1 = wmma16(frag_a(Kc, HD_, 16, 0),  qb0, p1);
  p1 = wmma16(frag_a(Kc, HD_, 16, 32), qb1, p1);

  float mx = -1e30f;
#pragma unroll
  for (int r = 0; r < 8; ++r) {
    const int ka = k0 + r + hi * 8;
    p0[r] = (ka      <= qidx) ? p0[r] * 0.125f : -10000.f;
    p1[r] = (ka + 16 <= qidx) ? p1[r] * 0.125f : -10000.f;
    mx = fmaxf(mx, fmaxf(p0[r], p1[r]));
  }
  mx = fmaxf(mx, __shfl_xor(mx, 16, 32));          // merge key halves
  const float m_new = fmaxf(m_run, mx);
  const float alpha = __expf(m_run - m_new);

  float sum = 0.f;
  v8h w0, w1;
#pragma unroll
  for (int r = 0; r < 8; ++r) {
    const float e0 = __expf(p0[r] - m_new);
    const float e1 = __expf(p1[r] - m_new);
    sum += e0 + e1;
    w0[r] = (h16)e0;
    w1[r] = (h16)e1;
  }
  sum += __shfl_xor(sum, 16, 32);
  l_run = l_run * alpha + sum;
  m_run = m_new;

  if (hi == 0) bb[n] = alpha;                      // per-query alpha
  {                                                // P^T[query][key] chunk
    h16* pw = pw_base + n * 32;
    *(v8h*)(pw + hi * 8)      = w0;
    *(v8h*)(pw + 16 + hi * 8) = w1;
  }
  {                                                // rescale O rows by alpha
    const float* ap = bb + hi * 8;
#pragma unroll
    for (int r = 0; r < 8; ++r) {
      const float a = ap[r];
      o0[r] *= a; o1[r] *= a; o2[r] *= a; o3[r] *= a;
    }
  }
  v16h af;                                         // P^T as A-fragment
  {
    const h16* p = pw_base + (lane & 15) * 32 + hi * 8;
    v8h lo = *(const v8h*)p;
    v8h hh = *(const v8h*)(p + 16);
#pragma unroll
    for (int i = 0; i < 8; ++i) { af[i] = lo[i]; af[i + 8] = hh[i]; }
  }
  o0 = wmma16(af, frag_b(Vc, 32, 0, 0),  o0);
  o1 = wmma16(af, frag_b(Vc, 32, 0, 16), o1);
  o2 = wmma16(af, frag_b(Vc, 32, 0, 32), o2);
  o3 = wmma16(af, frag_b(Vc, 32, 0, 48), o3);
}

// ---------------------------------------------------------------------------
// Causal flash attention. Block = 128 queries of one (b,h); uniform (even)
// chunk count so the 8 waves cooperatively double-buffer K (32x64) and
// V^T (64x32) tiles in LDS via async copies; chunk loop unrolled x2 so the
// LDS buffer selection is compile-time constant.
// ---------------------------------------------------------------------------
__global__ __launch_bounds__(256) void flash_attn(const h16* __restrict__ Qh,
                                                  const h16* __restrict__ Kh,
                                                  const h16* __restrict__ Vt,
                                                  h16* __restrict__ Oh) {
  __shared__ __align__(16) h16 KtL[2][32 * 64];    // [key][feat]
  __shared__ __align__(16) h16 VtL[2][64 * 32];    // [feat][key]
  __shared__ __align__(16) h16 lds_p[8][16 * 32];  // per-wave P^T chunk
  __shared__ float lds_b[8][16];                   // per-wave row broadcast

  const int t    = threadIdx.x;
  const int lane = t & 31;
  const int w    = t >> 5;
  const int n    = lane & 15;
  const int hi   = lane >> 4;
  const int qblk = blockIdx.x % (S_ / 128);
  const int bh   = blockIdx.x / (S_ / 128);
  const int b = bh / H_, h = bh % H_;
  const int q0 = qblk * 128 + w * 16;

  const h16* Qb = Qh + (size_t)bh * S_ * HD_;
  const h16* Kb = Kh + (size_t)bh * S_ * HD_;
  const h16* Vb = Vt + (size_t)bh * HD_ * S_;

  const v16h qb0 = frag_b(Qb, HD_, 0,  q0);
  const v16h qb1 = frag_b(Qb, HD_, 32, q0);

  v8f o0 = {}, o1 = {}, o2 = {}, o3 = {};
  float m_run = -1e30f, l_run = 0.f;
  const int qidx   = q0 + n;
  const int nchunk = 4 * qblk + 4;                 // uniform & even

  const int ve = t >> 2, vp = (t & 3) * 8;         // V staging: 64 rows x 4x16B
  h16* pw = &lds_p[w][0];
  float* bb = &lds_b[w][0];

  async_b128(Kb + (size_t)t * 8, &KtL[0][t * 8]);
  async_b128(Vb + (size_t)ve * S_ + vp, &VtL[0][ve * 32 + vp]);

  for (int c = 0; c < nchunk; c += 2) {
    // ---- phase A: compute chunk c from buf0, prefetch c+1 -> buf1 (always)
    {
      const int k1 = (c + 1) * 32;
      async_b128(Kb + (size_t)k1 * HD_ + t * 8, &KtL[1][t * 8]);
      async_b128(Vb + (size_t)ve * S_ + k1 + vp, &VtL[1][ve * 32 + vp]);
      wait_asynccnt<2>();
    }
    __syncthreads();
    flash_chunk(&KtL[0][0], &VtL[0][0], c * 32, qb0, qb1, qidx, hi, n,
                pw, bb, o0, o1, o2, o3, m_run, l_run);
    __syncthreads();
    // ---- phase B: compute chunk c+1 from buf1, prefetch c+2 -> buf0
    if (c + 2 < nchunk) {
      const int k2 = (c + 2) * 32;
      async_b128(Kb + (size_t)k2 * HD_ + t * 8, &KtL[0][t * 8]);
      async_b128(Vb + (size_t)ve * S_ + k2 + vp, &VtL[0][ve * 32 + vp]);
      wait_asynccnt<2>();
    } else {
      wait_asynccnt<0>();
    }
    __syncthreads();
    flash_chunk(&KtL[1][0], &VtL[1][0], (c + 1) * 32, qb0, qb1, qidx, hi, n,
                pw, bb, o0, o1, o2, o3, m_run, l_run);
    __syncthreads();
  }

  // normalize by 1/l per query row and emit f16 [token][D] for the out-proj
  if (hi == 0) bb[n] = 1.0f / l_run;
  const float* rp = bb + hi * 8;
  h16* outp = Oh + ((size_t)(b * S_ + q0 + hi * 8)) * D_ + h * HD_;
#pragma unroll
  for (int r = 0; r < 8; ++r) {
    const float s = rp[r];
    h16* row = outp + (size_t)r * D_;
    row[n]      = (h16)(o0[r] * s);
    row[16 + n] = (h16)(o1[r] * s);
    row[32 + n] = (h16)(o2[r] * s);
    row[48 + n] = (h16)(o3[r] * s);
  }
}

// ---------------------------------------------------------------------------
// Output projection: [4096 x 1024] x [1024 x 1024] + bias -> fp32 result.
// Same block-cooperative LDS weight staging as qkv_gemm.
// ---------------------------------------------------------------------------
__global__ __launch_bounds__(256) void out_gemm(const h16* __restrict__ Oh,
                                                const h16* __restrict__ Wt,
                                                const float* __restrict__ bias,
                                                float* __restrict__ out) {
  __shared__ __align__(16) h16 BtL[2][64 * 32];
  const int t    = threadIdx.x;
  const int lane = t & 31;
  const int w    = t >> 5;
  const int NB   = D_ / 64;                        // 16 column panels
  const int blkM = blockIdx.x / NB, blkN = blockIdx.x % NB;
  const int row0 = blkM * 256 + w * 32, col0 = blkN * 64;
  const int n = lane & 15, hi = lane >> 4;
  const int rr = t >> 2, piece = (t & 3) * 8;
  const h16* Wrow = Wt + (size_t)(col0 + rr) * D_ + piece;

  async_b128(Wrow, &BtL[0][rr * 32 + piece]);

  v8f acc[8] = {};
  for (int kk = 0; kk < D_; kk += 64) {
    async_b128(Wrow + (kk + 32), &BtL[1][rr * 32 + piece]);
    wait_asynccnt<1>();
    __syncthreads();
    __builtin_prefetch(Oh + (size_t)(row0 + (lane & 15)) * D_ + kk + 64, 0, 3);
    gemm_phase(Oh, row0, kk, &BtL[0][0], acc);
    __syncthreads();
    if (kk + 64 < D_) {
      async_b128(Wrow + (kk + 64), &BtL[0][rr * 32 + piece]);
      wait_asynccnt<1>();
    } else {
      wait_asynccnt<0>();
    }
    __syncthreads();
    gemm_phase(Oh, row0, kk + 32, &BtL[1][0], acc);
    __syncthreads();
  }

#pragma unroll
  for (int g = 0; g < 2; ++g) {
    const int rbase = row0 + g * 16 + hi * 8;
#pragma unroll
    for (int f = 0; f < 4; ++f) {
      const int col  = col0 + f * 16 + n;
      const float bv = bias[col];
      float* dst = out + (size_t)rbase * D_ + col;
#pragma unroll
      for (int r = 0; r < 8; ++r) dst[(size_t)r * D_] = acc[g * 4 + f][r] + bv;
    }
  }
}

// ---------------------------------------------------------------------------
extern "C" void kernel_launch(void* const* d_in, const int* in_sizes, int n_in,
                              void* d_out, int out_size, void* d_ws, size_t ws_size,
                              hipStream_t stream) {
  const float* x     = (const float*)d_in[0];
  const float* W_qkv = (const float*)d_in[1];
  const float* b_qkv = (const float*)d_in[2];
  const float* W_out = (const float*)d_in[3];
  const float* b_out = (const float*)d_in[4];
  float* out = (float*)d_out;

  // workspace carve-up (f16), total ~48 MB
  h16* p   = (h16*)d_ws;
  h16* xh  = p; p += (size_t)M_ * D_;              // 4096x1024
  h16* Wtq = p; p += (size_t)3 * D_ * D_;          // [3072][1024]
  h16* Wto = p; p += (size_t)D_ * D_;              // [1024][1024]
  h16* Qh  = p; p += (size_t)B_ * H_ * S_ * HD_;   // [b,h,s,e]
  h16* Kh  = p; p += (size_t)B_ * H_ * S_ * HD_;   // [b,h,s,e]
  h16* Vt  = p; p += (size_t)B_ * H_ * HD_ * S_;   // [b,h,e,s]
  h16* Oh  = p;                                    // 4096x1024 attn out

  cvt_f16<<<(M_ * D_) / 256, 256, 0, stream>>>(x, xh, M_ * D_);
  transpose_cvt<<<(3 * D_ * D_) / 256, 256, 0, stream>>>(W_qkv, Wtq, D_, 3 * D_);
  transpose_cvt<<<(D_ * D_) / 256, 256, 0, stream>>>(W_out, Wto, D_, D_);

  qkv_gemm<<<16 * 48, 256, 0, stream>>>(xh, Wtq, b_qkv, Qh, Kh, Vt);
  flash_attn<<<B_ * H_ * (S_ / 128), 256, 0, stream>>>(Qh, Kh, Vt, Oh);
  out_gemm<<<16 * 16, 256, 0, stream>>>(Oh, Wto, b_out, out);
}
